// ContextualBiasAttention_7739531067591
// MI455X (gfx1250) — matrix operand order repaired and checked by reading, not compiled, not run on hardware
//
#include <hip/hip_runtime.h>
#include <hip/hip_bf16.h>
#include <math.h>

typedef __attribute__((ext_vector_type(16))) _Float16 v16h;
typedef __attribute__((ext_vector_type(8)))  float    v8f;

constexpr int B_  = 2;
constexpr int S_  = 2048;
constexpr int H_  = 1024;
constexpr int NH_ = 16;
constexpr int HD_ = 64;
constexpr int PST = 2052;   // padded LDS row stride (floats) to break bank conflicts

#define DEVINL __device__ __forceinline__

DEVINL v8f wmma_f16(v16h a, v16h b, v8f c) {
    // D = A(16x32 f16) * B(32x16 f16) + C(16x16 f32)
    return __builtin_amdgcn_wmma_f32_16x16x32_f16(
        /*neg_a=*/false, a, /*neg_b=*/false, b,
        /*c_mod=*/(short)0, c, /*reuse_a=*/false, /*reuse_b=*/false);
}

// A fragment: 16x32 (MxK) f16 from row-major src (ld in elements).
// Lanes 0-15: row M=lane, K 0..15; lanes 16-31: same rows, K 16..31.
DEVINL v16h load_a_frag(const _Float16* __restrict__ src, int ld) {
    const int lane = threadIdx.x & 31;
    const int row  = lane & 15;
    const int kb   = (lane >> 4) << 4;          // 0 or 16
    const _Float16* p = src + (size_t)row * ld + kb;
    v16h a;
#pragma unroll
    for (int v = 0; v < 4; ++v) { a[2*v]   = p[2*v];      a[2*v+1]   = p[2*v+1]; }
#pragma unroll
    for (int v = 0; v < 4; ++v) { a[8+2*v] = p[8+2*v];   a[8+2*v+1] = p[8+2*v+1]; }
    return a;
}

// Same A layout but sourced from f32 LDS (probabilities), converted to f16.
DEVINL v16h load_a_frag_lds_f32(const float* __restrict__ p, int ld, int col0) {
    const int lane = threadIdx.x & 31;
    const int row  = lane & 15;
    const int kb   = (lane >> 4) << 4;
    const float* q = p + row * ld + col0 + kb;
    v16h a;
#pragma unroll
    for (int v = 0; v < 4; ++v) { a[2*v]   = (_Float16)q[2*v];    a[2*v+1]   = (_Float16)q[2*v+1]; }
#pragma unroll
    for (int v = 0; v < 4; ++v) { a[8+2*v] = (_Float16)q[8+2*v]; a[8+2*v+1] = (_Float16)q[8+2*v+1]; }
    return a;
}

// B fragment (32x16, KxN) where B(k,n) = W[n][k], W row-major [N x K] (i.e. a W^T GEMM).
// Lanes 0-15 hold K=0..15 (VGPR v half h -> K=2v+h), lanes 16-31 hold K=16..31; lane&15 = N.
DEVINL v16h load_bt_frag(const _Float16* __restrict__ W, int ld) {
    const int lane = threadIdx.x & 31;
    const int n    = lane & 15;
    const int kb   = (lane >> 4) << 4;          // 0 or 16
    const _Float16* p = W + (size_t)n * ld + kb; // 32 contiguous bytes per lane
    v16h b;
#pragma unroll
    for (int i = 0; i < 16; ++i) b[i] = p[i];
    return b;
}

// B fragment (32x16) from row-major B [K x N]: B(k,n) = src[k*ld + n].
DEVINL v16h load_b_frag(const _Float16* __restrict__ src, int ld) {
    const int lane = threadIdx.x & 31;
    const int n    = lane & 15;
    const int kb   = (lane >> 4) << 4;
    v16h b;
#pragma unroll
    for (int i = 0; i < 16; ++i) b[i] = src[(size_t)(kb + i) * ld + n];
    return b;
}

// C/D 16x16 f32 store: VGPR r holds M=r (lanes 0-15) / M=r+8 (lanes 16-31); N = lane&15.
DEVINL void store_c_f32(float* __restrict__ dst, int ld, v8f c) {
    const int lane = threadIdx.x & 31;
    const int col  = lane & 15;
    const int rb   = (lane >> 4) << 3;
#pragma unroll
    for (int r = 0; r < 8; ++r) dst[(size_t)(rb + r) * ld + col] = c[r];
}

// ---------------------------------------------------------------------------
__global__ void cvt_f32_f16(const float* __restrict__ src, _Float16* __restrict__ dst, int n) {
    for (int i = blockIdx.x * blockDim.x + threadIdx.x; i < n; i += gridDim.x * blockDim.x)
        dst[i] = (_Float16)src[i];
}

// ---------------------------------------------------------------------------
// QKV projection + RoPE.  grid = (heads=16, Mtiles=256), block = 128 (4 waves).
// Each block produces a 16(row) x 64(one head) tile of Q, K, V.
__global__ void __launch_bounds__(128)
qkv_rope_kernel(const _Float16* __restrict__ xh,
                const _Float16* __restrict__ Wqh,
                const _Float16* __restrict__ Wkh,
                const _Float16* __restrict__ Wvh,
                _Float16* __restrict__ Qr,
                _Float16* __restrict__ Kr,
                _Float16* __restrict__ Vh) {
    __shared__ float lq[16*64], lk[16*64], lv[16*64];
    const int h    = blockIdx.x;
    const int m0   = blockIdx.y * 16;        // row in flattened (B*S)
    const int b    = m0 >> 11;               // /S
    const int pos0 = m0 & (S_ - 1);
    const int wave = threadIdx.x >> 5;
    const int n0   = h * HD_ + wave * 16;    // output feature base

    v8f aq = {}, ak = {}, av = {};
    const _Float16* xrow = xh + (size_t)m0 * H_;
    for (int kc = 0; kc < H_; kc += 32) {
        if (kc + 32 < H_) __builtin_prefetch(xrow + kc + 32, 0, 1);
        v16h a  = load_a_frag(xrow + kc, H_);
        v16h bq = load_bt_frag(Wqh + (size_t)n0 * H_ + kc, H_);
        v16h bk = load_bt_frag(Wkh + (size_t)n0 * H_ + kc, H_);
        v16h bv = load_bt_frag(Wvh + (size_t)n0 * H_ + kc, H_);
        aq = wmma_f16(a, bq, aq);
        ak = wmma_f16(a, bk, ak);
        av = wmma_f16(a, bv, av);
    }
    store_c_f32(lq + wave * 16, 64, aq);
    store_c_f32(lk + wave * 16, 64, ak);
    store_c_f32(lv + wave * 16, 64, av);
    __syncthreads();

    constexpr float LN1E4_OVER_32 = 0.28779924046f;   // ln(10000)/32
#pragma unroll
    for (int i = 0; i < 8; ++i) {
        const int e   = threadIdx.x + i * 128;        // 16x64 elements
        const int row = e >> 6;
        const int d   = e & 63;
        const int pos = pos0 + row;
        const int j   = d & 31;
        const float invf = __expf(-(float)j * LN1E4_OVER_32);
        float sn, cs;
        __sincosf((float)pos * invf, &sn, &cs);
        const float qv   = lq[row * 64 + d];
        const float kv   = lk[row * 64 + d];
        const float qrot = (d < 32) ? -lq[row * 64 + d + 32] : lq[row * 64 + d - 32];
        const float krot = (d < 32) ? -lk[row * 64 + d + 32] : lk[row * 64 + d - 32];
        const size_t idx = ((size_t)(b * NH_ + h) * S_ + pos) * HD_ + d;
        Qr[idx] = (_Float16)(qv * cs + qrot * sn);
        Kr[idx] = (_Float16)(kv * cs + krot * sn);
        Vh[idx] = (_Float16)lv[row * 64 + d];
    }
}

// ---------------------------------------------------------------------------
// Fused scores + bias + mask + softmax + attn@V for 16 query rows.
// grid = (qtiles=128, b*h=32), block = 128 (4 waves).
// Dynamic LDS: 16 rows x PST f32 (131 KB) holds the full score/prob panel.
__global__ void __launch_bounds__(128)
attn_kernel(const _Float16* __restrict__ Qr,
            const _Float16* __restrict__ Kr,
            const _Float16* __restrict__ Vh,
            const float* __restrict__ bias,
            const int* __restrict__ amask,
            float* __restrict__ attnOut,
            _Float16* __restrict__ ctx) {
    extern __shared__ float probs[];
    const int qt   = blockIdx.x;
    const int bh   = blockIdx.y;
    const int b    = bh >> 4;
    const int h    = bh & 15;
    const int q0   = qt * 16;
    const int wave = threadIdx.x >> 5;
    const int lane = threadIdx.x & 31;
    const int klen   = q0 + 16;                 // causal: keys 0..q0+15
    const int padEnd = (klen + 31) & ~31;       // round up to K-chunk of 32

    const _Float16* Qbase = Qr + ((size_t)bh * S_ + q0) * HD_;
    const _Float16* Kbase = Kr + (size_t)bh * S_ * HD_;

    // Phase 1: scores = QK^T/8 + bias, masked, into LDS (skip tiles above diagonal)
    for (int kt = wave; kt <= qt; kt += 4) {
        v8f acc = {};
#pragma unroll
        for (int kc = 0; kc < HD_; kc += 32) {
            v16h a  = load_a_frag(Qbase + kc, HD_);
            v16h bt = load_bt_frag(Kbase + (size_t)kt * 16 * HD_ + kc, HD_);
            acc = wmma_f16(a, bt, acc);
        }
        const int col  = lane & 15;
        const int rb   = (lane >> 4) << 3;
        const int kcol = kt * 16 + col;
        const int am   = amask[(size_t)b * S_ + kcol];
#pragma unroll
        for (int r = 0; r < 8; ++r) {
            const int row  = rb + r;
            const int grow = q0 + row;
            float sc = acc[r] * 0.125f + bias[(size_t)grow * S_ + kcol];
            if (am == 0 || kcol > grow) sc = -INFINITY;
            probs[row * PST + kcol] = sc;
        }
    }
    // zero the K-chunk padding [klen, padEnd) so phase-3 fragments are clean
    for (int idx = threadIdx.x; idx < 16 * (padEnd - klen); idx += 128) {
        const int r = idx >> 4;
        const int c = klen + (idx & 15);
        probs[r * PST + c] = 0.0f;
    }
    __syncthreads();

    // Phase 2: row softmax (4 rows per wave, wave32 shuffle reductions) + stream attn out
    for (int i = 0; i < 4; ++i) {
        const int row = wave * 4 + i;
        float* prow = probs + row * PST;
        float mx = -INFINITY;
        for (int c = lane; c < klen; c += 32) mx = fmaxf(mx, prow[c]);
#pragma unroll
        for (int off = 16; off > 0; off >>= 1) mx = fmaxf(mx, __shfl_xor(mx, off, 32));
        float sum = 0.0f;
        for (int c = lane; c < klen; c += 32) {
            float e = __expf(prow[c] - mx);
            prow[c] = e;
            sum += e;
        }
#pragma unroll
        for (int off = 16; off > 0; off >>= 1) sum += __shfl_xor(sum, off, 32);
        const float inv = 1.0f / sum;
        float* arow = attnOut + ((size_t)bh * S_ + q0 + row) * S_;
        for (int c = lane; c < klen; c += 32) {
            float p = prow[c] * inv;
            prow[c] = p;
            arow[c] = p;
        }
        for (int c = klen + lane; c < S_; c += 32) arow[c] = 0.0f;  // causal zeros
    }
    __syncthreads();

    // Phase 3: out = probs @ V, probs fed straight from LDS (f32->f16), one 16-col slice/wave
    const int n0 = wave * 16;
    v8f acc = {};
    for (int c32 = 0; c32 < padEnd; c32 += 32) {
        v16h a  = load_a_frag_lds_f32(probs, PST, c32);
        v16h bf = load_b_frag(Vh + ((size_t)bh * S_ + c32) * HD_ + n0, HD_);
        acc = wmma_f16(a, bf, acc);
    }
    const int col = lane & 15;
    const int rb  = (lane >> 4) << 3;
#pragma unroll
    for (int r = 0; r < 8; ++r) {
        const int row = rb + r;
        ctx[((size_t)(b * S_ + q0 + row)) * H_ + h * HD_ + n0 + col] = (_Float16)acc[r];
    }
}

// ---------------------------------------------------------------------------
// Output projection: out = ctx @ Wo^T.  grid = (16, 256), block = 128.
__global__ void __launch_bounds__(128)
oproj_kernel(const _Float16* __restrict__ ctx,
             const _Float16* __restrict__ Woh,
             float* __restrict__ out) {
    const int m0   = blockIdx.y * 16;
    const int wave = threadIdx.x >> 5;
    const int n0   = blockIdx.x * 64 + wave * 16;
    v8f acc = {};
    const _Float16* crow = ctx + (size_t)m0 * H_;
    for (int kc = 0; kc < H_; kc += 32) {
        if (kc + 32 < H_) __builtin_prefetch(crow + kc + 32, 0, 1);
        v16h a  = load_a_frag(crow + kc, H_);
        v16h bt = load_bt_frag(Woh + (size_t)n0 * H_ + kc, H_);
        acc = wmma_f16(a, bt, acc);
    }
    store_c_f32(out + (size_t)m0 * H_ + n0, H_, acc);
}

// ---------------------------------------------------------------------------
extern "C" void kernel_launch(void* const* d_in, const int* in_sizes, int n_in,
                              void* d_out, int out_size, void* d_ws, size_t ws_size,
                              hipStream_t stream) {
    (void)in_sizes; (void)n_in; (void)out_size; (void)ws_size;
    const float* x    = (const float*)d_in[0];
    const int*   am   = (const int*)d_in[1];
    const float* Wq   = (const float*)d_in[2];
    const float* Wk   = (const float*)d_in[3];
    const float* Wv   = (const float*)d_in[4];
    const float* Wo   = (const float*)d_in[5];
    const float* bias = (const float*)d_in[6];

    const size_t NX = (size_t)B_ * S_ * H_;   // 4,194,304
    const size_t NW = (size_t)H_ * H_;        // 1,048,576

    _Float16* ws  = (_Float16*)d_ws;
    size_t off = 0;
    _Float16* xh  = ws + off; off += NX;
    _Float16* Wqh = ws + off; off += NW;
    _Float16* Wkh = ws + off; off += NW;
    _Float16* Wvh = ws + off; off += NW;
    _Float16* Woh = ws + off; off += NW;
    _Float16* Qr  = ws + off; off += NX;
    _Float16* Kr  = ws + off; off += NX;
    _Float16* Vh  = ws + off; off += NX;
    _Float16* ctx = ws + off; off += NX;      // total ~48 MB of f16 scratch

    float* outProj = (float*)d_out;
    float* attnOut = outProj + NX;

    cvt_f32_f16<<<2048, 256, 0, stream>>>(x,  xh,  (int)NX);
    cvt_f32_f16<<<1024, 256, 0, stream>>>(Wq, Wqh, (int)NW);
    cvt_f32_f16<<<1024, 256, 0, stream>>>(Wk, Wkh, (int)NW);
    cvt_f32_f16<<<1024, 256, 0, stream>>>(Wv, Wvh, (int)NW);
    cvt_f32_f16<<<1024, 256, 0, stream>>>(Wo, Woh, (int)NW);

    qkv_rope_kernel<<<dim3(NH_, (B_ * S_) / 16), 128, 0, stream>>>(
        xh, Wqh, Wkh, Wvh, Qr, Kr, Vh);

    const int shmem = 16 * PST * (int)sizeof(float);   // 131,328 B < 320 KB LDS
    hipFuncSetAttribute((const void*)attn_kernel,
                        hipFuncAttributeMaxDynamicSharedMemorySize, shmem);
    attn_kernel<<<dim3(S_ / 16, B_ * NH_), 128, shmem, stream>>>(
        Qr, Kr, Vh, bias, am, attnOut, ctx);

    oproj_kernel<<<dim3(H_ / 64, (B_ * S_) / 16), 128, 0, stream>>>(
        ctx, Woh, outProj);
}
